// MoeLayer_78297253806415
// MI455X (gfx1250) — compile-verified, matrix operand order checked by reading, not run on hardware
//
#include <hip/hip_runtime.h>
#include <cstdint>

#define E_    8
#define KTOP  4
#define B_    4
#define S_    1024
#define D_    1024
#define H_    2730
#define HPAD  2752          // 86 * 32 : zero-padded K/N for uniform tiles
#define T_    (B_*S_)       // 4096 tokens

typedef __attribute__((ext_vector_type(16))) __bf16 v16bf;
typedef __attribute__((ext_vector_type(8)))  float  v8f;
typedef int v4i_ __attribute__((vector_size(16)));   // matches builtin param type

union FragBF { uint4 u[2]; v16bf v; };   // 32B WMMA A/B operand
union Pack8  { __bf16 h[8]; uint4 u; };  // 16B = 8 bf16

#ifndef __has_builtin
#define __has_builtin(x) 0
#endif
#if __has_builtin(__builtin_amdgcn_global_load_async_to_lds_b128) && \
    __has_builtin(__builtin_amdgcn_s_wait_asynccnt)
#define ASYNC_LDS 1
#else
#define ASYNC_LDS 0
#endif

// 16-byte global -> LDS copy: async DMA path on CDNA5, fallback = direct copy
__device__ __forceinline__ void cp8(__bf16* dst_lds, const __bf16* src) {
#if ASYNC_LDS
  __builtin_amdgcn_global_load_async_to_lds_b128(
      (__attribute__((address_space(1))) v4i_*)(uintptr_t)src,
      (__attribute__((address_space(3))) v4i_*)(uintptr_t)dst_lds,
      0, 0);
#else
  *(uint4*)dst_lds = *(const uint4*)src;
#endif
}

#if ASYNC_LDS
#define WAIT_ASYNC(n) __builtin_amdgcn_s_wait_asynccnt(n)
#else
#define WAIT_ASYNC(n)
#endif

// ---------------------------------------------------------------------------
// Kernel 1: gating — gate = x@Wg^T + bias, top-4, masked softmax -> weights
// ---------------------------------------------------------------------------
__global__ void gate_topk_kernel(const float* __restrict__ x,
                                 const float* __restrict__ Wg,
                                 const float* __restrict__ rbias,
                                 float* __restrict__ weights) {
  const int wid  = threadIdx.x >> 5;
  const int lane = threadIdx.x & 31;
  const int t = blockIdx.x * (blockDim.x >> 5) + wid;
  if (t >= T_) return;

  float acc[E_];
#pragma unroll
  for (int e = 0; e < E_; ++e) acc[e] = 0.f;

  const float* xr = x + (size_t)t * D_;
  for (int d = lane; d < D_; d += 32) {
    float xv = xr[d];
#pragma unroll
    for (int e = 0; e < E_; ++e) acc[e] += xv * Wg[e * D_ + d];
  }
#pragma unroll
  for (int e = 0; e < E_; ++e) {
#pragma unroll
    for (int off = 16; off >= 1; off >>= 1)
      acc[e] += __shfl_xor(acc[e], off, 32);
  }

  if (lane == 0) {
    float g[E_]; bool sel[E_];
#pragma unroll
    for (int e = 0; e < E_; ++e) { g[e] = acc[e] + rbias[e]; sel[e] = false; }
    for (int k = 0; k < KTOP; ++k) {
      int best = 0; float bv = -3.4e38f;
      for (int e = 0; e < E_; ++e)
        if (!sel[e] && g[e] > bv) { bv = g[e]; best = e; }
      sel[best] = true;
    }
    float m = -3.4e38f;
    for (int e = 0; e < E_; ++e) if (sel[e] && g[e] > m) m = g[e];
    float p[E_]; float s = 0.f;
    for (int e = 0; e < E_; ++e) { p[e] = sel[e] ? __expf(g[e] - m) : 0.f; s += p[e]; }
    float inv = 1.f / s;
    for (int e = 0; e < E_; ++e) weights[(size_t)t * E_ + e] = p[e] * inv;
  }
}

// ---------------------------------------------------------------------------
// Kernel 2a: fp32 -> bf16, pad extra ROWS with zeros (also used for x: no pad)
// ---------------------------------------------------------------------------
__global__ void cvt_rows_pad_kernel(const float* __restrict__ src,
                                    __bf16* __restrict__ dst,
                                    int rows_src, int rows_dst, int cols) {
  long i = (long)(blockIdx.x * blockDim.x + threadIdx.x) * 8;
  if (i >= (long)rows_dst * cols) return;
  int row = (int)(i / cols);
  int col = (int)(i % cols);
  Pack8 p;
  if (row < rows_src) {
    const float4* s = (const float4*)(src + (size_t)row * cols + col);
    float4 a = s[0], b = s[1];
    p.h[0]=(__bf16)a.x; p.h[1]=(__bf16)a.y; p.h[2]=(__bf16)a.z; p.h[3]=(__bf16)a.w;
    p.h[4]=(__bf16)b.x; p.h[5]=(__bf16)b.y; p.h[6]=(__bf16)b.z; p.h[7]=(__bf16)b.w;
  } else {
    p.u = make_uint4(0, 0, 0, 0);
  }
  *(uint4*)(dst + i) = p.u;
}

// ---------------------------------------------------------------------------
// Kernel 2b: fp32 -> bf16, pad extra COLUMNS with zeros (W3: [D,H]->[D,HPAD])
// ---------------------------------------------------------------------------
__global__ void cvt_cols_pad_kernel(const float* __restrict__ src,
                                    __bf16* __restrict__ dst,
                                    int rows, int cols_src, int cols_dst) {
  long i = (long)(blockIdx.x * blockDim.x + threadIdx.x) * 8;
  if (i >= (long)rows * cols_dst) return;
  int row = (int)(i / cols_dst);
  int col = (int)(i % cols_dst);
  Pack8 p;
#pragma unroll
  for (int j = 0; j < 8; ++j) {
    int c = col + j;
    p.h[j] = (c < cols_src) ? (__bf16)src[(size_t)row * cols_src + c] : (__bf16)0.f;
  }
  *(uint4*)(dst + i) = p.u;
}

// ---------------------------------------------------------------------------
// Kernel 3: act = silu(x@W1^T) * (x@W2^T)   (all operands bf16, zero-padded)
// 128 tokens x 64 h-cols per block; double-buffered async LDS staging;
// 8 waves x (2x2 tiles) x 2 matrices = 8 v_wmma per k-step
// ---------------------------------------------------------------------------
__global__ __launch_bounds__(256)
void ffn_act_kernel(const __bf16* __restrict__ xb,
                    const __bf16* __restrict__ w1b,   // [HPAD, D] zero-padded rows
                    const __bf16* __restrict__ w2b,
                    __bf16* __restrict__ act) {       // [T, HPAD]
  __shared__ __bf16 sA [2][128 * 32];
  __shared__ __bf16 sB1[2][ 64 * 32];
  __shared__ __bf16 sB2[2][ 64 * 32];

  const int tid  = threadIdx.x;
  const int lane = tid & 31;
  const int wid  = tid >> 5;
  const int wm   = wid & 3;
  const int wn   = wid >> 2;
  const int m0   = blockIdx.x * 128;
  const int n0   = blockIdx.y * 64;

  const v8f vzero = {0.f,0.f,0.f,0.f,0.f,0.f,0.f,0.f};
  v8f acc1[2][2], acc2[2][2];
#pragma unroll
  for (int i = 0; i < 2; ++i)
#pragma unroll
    for (int j = 0; j < 2; ++j) { acc1[i][j] = vzero; acc2[i][j] = vzero; }

  const int arow  = tid >> 1;          // 0..127
  const int ahalf = (tid & 1) * 16;    // 0 / 16
  const int br    = tid >> 2;          // 0..63
  const int bc    = (tid & 3) * 8;     // 0,8,16,24

  auto issue = [&](int kk, int buf) {  // 4 async b128 per thread per stage
    const __bf16* aSrc = xb + (size_t)(m0 + arow) * D_ + kk + ahalf;
    cp8(&sA[buf][arow * 32 + ahalf],     aSrc);
    cp8(&sA[buf][arow * 32 + ahalf + 8], aSrc + 8);
    cp8(&sB1[buf][br * 32 + bc], w1b + (size_t)(n0 + br) * D_ + kk + bc);
    cp8(&sB2[buf][br * 32 + bc], w2b + (size_t)(n0 + br) * D_ + kk + bc);
  };

  const int NK = D_ / 32;              // 32 k-steps
  issue(0, 0);
  issue(32, 1);

  for (int ks = 0; ks < NK; ++ks) {
    const int cur = ks & 1;
    if (ks + 1 < NK) { WAIT_ASYNC(4); } else { WAIT_ASYNC(0); }
    __syncthreads();

    const int klo = (lane >> 4) * 8;
    FragBF af[2], b1f[2], b2f[2];
#pragma unroll
    for (int mi = 0; mi < 2; ++mi) {
      const __bf16* base = &sA[cur][(wm * 32 + mi * 16 + (lane & 15)) * 32];
      af[mi].u[0] = *(const uint4*)(base + klo);
      af[mi].u[1] = *(const uint4*)(base + klo + 16);
    }
#pragma unroll
    for (int ni = 0; ni < 2; ++ni) {
      const __bf16* base1 = &sB1[cur][(wn * 32 + ni * 16 + (lane & 15)) * 32];
      const __bf16* base2 = &sB2[cur][(wn * 32 + ni * 16 + (lane & 15)) * 32];
      b1f[ni].u[0] = *(const uint4*)(base1 + klo);
      b1f[ni].u[1] = *(const uint4*)(base1 + klo + 16);
      b2f[ni].u[0] = *(const uint4*)(base2 + klo);
      b2f[ni].u[1] = *(const uint4*)(base2 + klo + 16);
    }
#pragma unroll
    for (int mi = 0; mi < 2; ++mi)
#pragma unroll
      for (int ni = 0; ni < 2; ++ni) {
        acc1[mi][ni] = __builtin_amdgcn_wmma_f32_16x16x32_bf16(
            false, af[mi].v, false, b1f[ni].v, (short)0, acc1[mi][ni], false, false);
        acc2[mi][ni] = __builtin_amdgcn_wmma_f32_16x16x32_bf16(
            false, af[mi].v, false, b2f[ni].v, (short)0, acc2[mi][ni], false, false);
      }
    __syncthreads();                       // everyone done reading buf `cur`
    if (ks + 2 < NK) issue((ks + 2) * 32, cur);
  }

#pragma unroll
  for (int mi = 0; mi < 2; ++mi)
#pragma unroll
    for (int ni = 0; ni < 2; ++ni) {
      const int nn = n0 + wn * 32 + ni * 16 + (lane & 15);
      const int mbase = m0 + wm * 32 + mi * 16 + (lane >> 4) * 8;
#pragma unroll
      for (int r = 0; r < 8; ++r) {
        float s1 = acc1[mi][ni][r];
        float s2 = acc2[mi][ni][r];
        float v  = s1 * (1.f / (1.f + __expf(-s1))) * s2;
        act[(size_t)(mbase + r) * HPAD + nn] = (nn < H_) ? (__bf16)v : (__bf16)0.f;
      }
    }
}

// ---------------------------------------------------------------------------
// Kernel 4: eo = act @ W3^T ; out = eo (shared, e<0) or out += w[t,e]*eo
// ---------------------------------------------------------------------------
__global__ __launch_bounds__(256)
void ffn_out_kernel(const __bf16* __restrict__ act,   // [T, HPAD]
                    const __bf16* __restrict__ w3b,   // [D, HPAD] zero-padded cols
                    const float* __restrict__ weights, int e,
                    float* __restrict__ out) {
  __shared__ __bf16 sA[2][128 * 32];
  __shared__ __bf16 sB[2][ 64 * 32];

  const int tid  = threadIdx.x;
  const int lane = tid & 31;
  const int wid  = tid >> 5;
  const int wm   = wid & 3;
  const int wn   = wid >> 2;
  const int m0   = blockIdx.x * 128;
  const int n0   = blockIdx.y * 64;

  const v8f vzero = {0.f,0.f,0.f,0.f,0.f,0.f,0.f,0.f};
  v8f acc[2][2];
#pragma unroll
  for (int i = 0; i < 2; ++i)
#pragma unroll
    for (int j = 0; j < 2; ++j) acc[i][j] = vzero;

  const int arow  = tid >> 1;
  const int ahalf = (tid & 1) * 16;
  const int br    = tid >> 2;
  const int bc    = (tid & 3) * 8;

  auto issue = [&](int kk, int buf) {  // 3 async b128 per thread per stage
    const __bf16* aSrc = act + (size_t)(m0 + arow) * HPAD + kk + ahalf;
    cp8(&sA[buf][arow * 32 + ahalf],     aSrc);
    cp8(&sA[buf][arow * 32 + ahalf + 8], aSrc + 8);
    cp8(&sB[buf][br * 32 + bc], w3b + (size_t)(n0 + br) * HPAD + kk + bc);
  };

  const int NK = HPAD / 32;            // 86 k-steps
  issue(0, 0);
  issue(32, 1);

  for (int ks = 0; ks < NK; ++ks) {
    const int cur = ks & 1;
    if (ks + 1 < NK) { WAIT_ASYNC(3); } else { WAIT_ASYNC(0); }
    __syncthreads();

    const int klo = (lane >> 4) * 8;
    FragBF af[2], bf[2];
#pragma unroll
    for (int mi = 0; mi < 2; ++mi) {
      const __bf16* base = &sA[cur][(wm * 32 + mi * 16 + (lane & 15)) * 32];
      af[mi].u[0] = *(const uint4*)(base + klo);
      af[mi].u[1] = *(const uint4*)(base + klo + 16);
    }
#pragma unroll
    for (int ni = 0; ni < 2; ++ni) {
      const __bf16* base = &sB[cur][(wn * 32 + ni * 16 + (lane & 15)) * 32];
      bf[ni].u[0] = *(const uint4*)(base + klo);
      bf[ni].u[1] = *(const uint4*)(base + klo + 16);
    }
#pragma unroll
    for (int mi = 0; mi < 2; ++mi)
#pragma unroll
      for (int ni = 0; ni < 2; ++ni)
        acc[mi][ni] = __builtin_amdgcn_wmma_f32_16x16x32_bf16(
            false, af[mi].v, false, bf[ni].v, (short)0, acc[mi][ni], false, false);
    __syncthreads();
    if (ks + 2 < NK) issue((ks + 2) * 32, cur);
  }

#pragma unroll
  for (int mi = 0; mi < 2; ++mi)
#pragma unroll
    for (int ni = 0; ni < 2; ++ni) {
      const int d = n0 + wn * 32 + ni * 16 + (lane & 15);
      const int mbase = m0 + wm * 32 + mi * 16 + (lane >> 4) * 8;
#pragma unroll
      for (int r = 0; r < 8; ++r) {
        const int t = mbase + r;
        const size_t idx = (size_t)t * D_ + d;
        float v = acc[mi][ni][r];
        if (e >= 0) out[idx] += weights[(size_t)t * E_ + e] * v;  // experts
        else        out[idx] = v;                                 // shared: overwrite
      }
    }
}

// ---------------------------------------------------------------------------
extern "C" void kernel_launch(void* const* d_in, const int* in_sizes, int n_in,
                              void* d_out, int out_size, void* d_ws, size_t ws_size,
                              hipStream_t stream) {
  const float* x   = (const float*)d_in[0];
  const float* Wg  = (const float*)d_in[1];
  const float* rb  = (const float*)d_in[2];
  const float* W1  = (const float*)d_in[3];
  const float* W2  = (const float*)d_in[4];
  const float* W3  = (const float*)d_in[5];
  const float* Ws1 = (const float*)d_in[6];
  const float* Ws2 = (const float*)d_in[7];
  const float* Ws3 = (const float*)d_in[8];
  float* out = (float*)d_out;

  // workspace layout (bytes)
  char* ws = (char*)d_ws;
  float*  weights = (float*)ws;                            // T*E f32        128 KB
  size_t  off = 131072;
  __bf16* xb  = (__bf16*)(ws + off); off += (size_t)T_ * D_ * 2;        // 8 MB
  __bf16* act = (__bf16*)(ws + off); off += (size_t)T_ * HPAD * 2;      // 22.5 MB
  __bf16* w1b = (__bf16*)(ws + off); off += (size_t)HPAD * D_ * 2;      // 5.6 MB
  __bf16* w2b = (__bf16*)(ws + off); off += (size_t)HPAD * D_ * 2;      // 5.6 MB
  __bf16* w3b = (__bf16*)(ws + off);                                    // 5.6 MB

  gate_topk_kernel<<<T_ / 8, 256, 0, stream>>>(x, Wg, rb, weights);

  // x -> bf16 (no padding: rows_src == rows_dst)
  cvt_rows_pad_kernel<<<((long)T_ * D_ / 8 + 255) / 256, 256, 0, stream>>>(
      x, xb, T_, T_, D_);

  dim3 g1(T_ / 128, HPAD / 64);   // 32 x 43
  dim3 g2(T_ / 128, D_ / 64);     // 32 x 16
  const int gw  = (int)(((long)HPAD * D_ / 8 + 255) / 256);   // W1/W2/W3 cvt grids

  // ---- shared expert first: fully overwrites d_out ----
  cvt_rows_pad_kernel<<<gw, 256, 0, stream>>>(Ws1, w1b, H_, HPAD, D_);
  cvt_rows_pad_kernel<<<gw, 256, 0, stream>>>(Ws2, w2b, H_, HPAD, D_);
  cvt_cols_pad_kernel<<<gw, 256, 0, stream>>>(Ws3, w3b, D_, H_, HPAD);
  ffn_act_kernel<<<g1, 256, 0, stream>>>(xb, w1b, w2b, act);
  ffn_out_kernel<<<g2, 256, 0, stream>>>(act, w3b, weights, -1, out);

  // ---- routed experts: accumulate with combine weights ----
  for (int e = 0; e < E_; ++e) {
    cvt_rows_pad_kernel<<<gw, 256, 0, stream>>>(W1 + (size_t)e * H_ * D_, w1b, H_, HPAD, D_);
    cvt_rows_pad_kernel<<<gw, 256, 0, stream>>>(W2 + (size_t)e * H_ * D_, w2b, H_, HPAD, D_);
    cvt_cols_pad_kernel<<<gw, 256, 0, stream>>>(W3 + (size_t)e * (size_t)D_ * H_, w3b, D_, H_, HPAD);
    ffn_act_kernel<<<g1, 256, 0, stream>>>(xb, w1b, w2b, act);
    ffn_out_kernel<<<g2, 256, 0, stream>>>(act, w3b, weights, e, out);
  }
}
